// SequenceModel_3487513444643
// MI455X (gfx1250) — matrix operand order repaired and checked by reading, not compiled, not run on hardware
//
#include <hip/hip_runtime.h>
#include <hip/hip_bf16.h>
#include <cstdint>
#include <cstddef>

typedef _Float16 f16;
typedef _Float16 v8h  __attribute__((ext_vector_type(8)));
typedef _Float16 v16h __attribute__((ext_vector_type(16)));
typedef float    v8f  __attribute__((ext_vector_type(8)));

#define B_      128
#define T_      64
#define TM1     63
#define V_      8192
#define DQ_     256
#define DS_     512
#define DQM_    256
#define KEXP    8
#define CTX_    64
#define MAXMEM  32
#define NPOS    64
#define XD_     513
#define XDP     544      /* 513 padded up to 17*32 */
#define DT_F    0.2f
#define DAMP_F  0.1f
#define SCALE_F 0.0625f  /* DQM^-0.5 */

// ---------------------------------------------------------------------------
// WMMA fragment loaders (CDNA5 16x16x32 f16 layouts, wave32)
// A: row-major MxK f16. lane holds row M = lane&15; K-halves at k0+8*(lane>>4), +16.
// B: stored transposed (NxK row-major). lane holds col N = lane&15; K-halves at
//    k0+8*(lane>>4), +16 (same order as A).
// ---------------------------------------------------------------------------
__device__ __forceinline__ v16h load_a_frag(const f16* __restrict__ A, int lda,
                                            int row0, int k0, int lane) {
  int half = lane >> 4, l = lane & 15;
  const f16* p = A + (size_t)(row0 + l) * lda + k0 + half * 8;
  v8h lo = *(const v8h*)p;
  v8h hi = *(const v8h*)(p + 16);
  v16h r;
#pragma unroll
  for (int i = 0; i < 8; i++) { r[i] = lo[i]; r[i + 8] = hi[i]; }
  return r;
}

__device__ __forceinline__ v16h load_b_frag(const f16* __restrict__ Bt, int ldb,
                                            int col0, int k0, int lane) {
  int half = lane >> 4, l = lane & 15;
  const f16* p = Bt + (size_t)(col0 + l) * ldb + k0 + half * 8;
  v8h lo = *(const v8h*)p;
  v8h hi = *(const v8h*)(p + 16);
  v16h r;
#pragma unroll
  for (int i = 0; i < 8; i++) { r[i] = lo[i]; r[i + 8] = hi[i]; }
  return r;
}

// ---------------------------------------------------------------------------
// Generic WMMA GEMM, register-blocked 16x64 per wave (4 accumulators; A frag
// reused across 4 B tiles). C[M,N] = act(A*Bt^T + bias). Requires N % 64 == 0.
// ---------------------------------------------------------------------------
__global__ void gemm_kernel(const f16* __restrict__ A, const f16* __restrict__ Bt,
                            const float* __restrict__ bias,
                            float* __restrict__ C, f16* __restrict__ Ch,
                            int M, int N, int Kd, int lda, int ldb, int ldc,
                            int act, int accumulate) {
  int wave = (int)((blockIdx.x * blockDim.x + threadIdx.x) >> 5);
  int lane = threadIdx.x & 31;
  int ntn = N >> 6;  // 64-wide N blocks
  int tm = wave / ntn, tn = wave % ntn;
  if (tm >= (M >> 4)) return;
  int row0 = tm << 4, col0 = tn << 6;
  v8f acc[4] = {{}, {}, {}, {}};
  for (int k0 = 0; k0 < Kd; k0 += 32) {
    v16h a = load_a_frag(A, lda, row0, k0, lane);
#pragma unroll
    for (int j = 0; j < 4; j++) {
      v16h b = load_b_frag(Bt, ldb, col0 + j * 16, k0, lane);
      acc[j] = __builtin_amdgcn_wmma_f32_16x16x32_f16(false, a, false, b,
                                                      (short)0, acc[j], false, false);
    }
  }
  int half = lane >> 4, l = lane & 15;
#pragma unroll
  for (int j = 0; j < 4; j++) {
    int col = col0 + j * 16 + l;
    float bv = bias ? bias[col] : 0.0f;
#pragma unroll
    for (int r = 0; r < 8; r++) {
      int row = row0 + r + half * 8;
      float v = acc[j][r] + bv;
      if (act) v = tanhf(v);
      size_t idx = (size_t)row * ldc + col;
      if (C)  C[idx] = accumulate ? (C[idx] + v) : v;
      if (Ch) Ch[idx] = (f16)v;
    }
  }
}

// ---------------------------------------------------------------------------
// Expert readout: out[b,v] = sum_k chart_w[b,k] * (x . Wro[k,v,:] + bias[k,v])
// 16x64 per wave; x tile staged in LDS (shared by all 8 waves of the block,
// re-read 8 experts x 17 k-steps). Top-2 sparsity: skip expert if every row
// in the 16-row tile has weight 0.
// ---------------------------------------------------------------------------
__global__ void readout_kernel(const f16* __restrict__ x_h, const f16* __restrict__ Wro,
                               const float* __restrict__ ro_bias,
                               const float* __restrict__ chart_w,
                               float* __restrict__ out, int ldc) {
  __shared__ __align__(32) f16 As[16 * XDP];  // 17.4 KB of the 320 KB WGP LDS
  int wave = (int)((blockIdx.x * blockDim.x + threadIdx.x) >> 5);
  int lane = threadIdx.x & 31;
  const int ntn = V_ >> 6;  // 128
  int tm = wave / ntn, tn = wave % ntn;  // tm uniform per block (128 % 8 == 0)
  int row0 = tm << 4, col0 = tn << 6;
  // cooperative stage of the 16 x XDP A tile
  for (int i = threadIdx.x; i < 16 * XDP; i += 256) {
    int r = i / XDP, c = i % XDP;
    As[i] = x_h[(size_t)(row0 + r) * XDP + c];
  }
  __syncthreads();
  int half = lane >> 4, l = lane & 15;
  float res[4][8];
#pragma unroll
  for (int j = 0; j < 4; j++)
#pragma unroll
    for (int r = 0; r < 8; r++) res[j][r] = 0.f;

  for (int k = 0; k < KEXP; k++) {
    float cw_probe = chart_w[(row0 + l) * KEXP + k];
    if (__builtin_amdgcn_ballot_w32(cw_probe != 0.0f) == 0u) continue;
    const f16* Bt = Wro + (size_t)k * V_ * XDP;
    v8f acc[4] = {{}, {}, {}, {}};
    for (int k0 = 0; k0 < XDP; k0 += 32) {
      // A fragment from LDS (row within tile is just l)
      const f16* pa = &As[l * XDP + k0 + half * 8];
      v8h lo = *(const v8h*)pa;
      v8h hi = *(const v8h*)(pa + 16);
      v16h a;
#pragma unroll
      for (int i = 0; i < 8; i++) { a[i] = lo[i]; a[i + 8] = hi[i]; }
#pragma unroll
      for (int j = 0; j < 4; j++) {
        v16h b = load_b_frag(Bt, XDP, col0 + j * 16, k0, lane);
        acc[j] = __builtin_amdgcn_wmma_f32_16x16x32_f16(false, a, false, b,
                                                        (short)0, acc[j], false, false);
      }
    }
#pragma unroll
    for (int j = 0; j < 4; j++) {
      int col = col0 + j * 16 + l;
      float bv = ro_bias[(size_t)k * V_ + col];
#pragma unroll
      for (int r = 0; r < 8; r++) {
        int row = row0 + r + half * 8;
        float cw = chart_w[row * KEXP + k];
        res[j][r] += cw * (acc[j][r] + bv);
      }
    }
  }
#pragma unroll
  for (int j = 0; j < 4; j++) {
    int col = col0 + j * 16 + l;
#pragma unroll
    for (int r = 0; r < 8; r++) {
      int row = row0 + r + half * 8;
      out[(size_t)row * ldc + col] = res[j][r];
    }
  }
}

// ---------------------------------------------------------------------------
// EVO dynamics: force = sum_k cw[b,k]*tanh(q@A[k]+b[k]) + drive; leapfrog.
// 16x64 per wave. Reads (q_h,q,p), writes (qn,pn) -> no WAR hazard.
// ---------------------------------------------------------------------------
__global__ void force_kernel(const f16* __restrict__ q_h, const float* __restrict__ q,
                             const float* __restrict__ p, const f16* __restrict__ dynAt,
                             const float* __restrict__ dyn_b,
                             const float* __restrict__ chart_w,
                             const float* __restrict__ drive,
                             float* __restrict__ qn, float* __restrict__ pn) {
  int wave = (int)((blockIdx.x * blockDim.x + threadIdx.x) >> 5);
  int lane = threadIdx.x & 31;
  const int ntn = DQ_ >> 6;  // 4
  int tm = wave / ntn, tn = wave % ntn;
  if (tm >= (B_ >> 4)) return;
  int row0 = tm << 4, col0 = tn << 6;
  int half = lane >> 4, l = lane & 15;
  float force[4][8];
#pragma unroll
  for (int j = 0; j < 4; j++)
#pragma unroll
    for (int r = 0; r < 8; r++) force[j][r] = 0.f;

  for (int k = 0; k < KEXP; k++) {
    float cw_probe = chart_w[(row0 + l) * KEXP + k];
    if (__builtin_amdgcn_ballot_w32(cw_probe != 0.0f) == 0u) continue;
    const f16* Bt = dynAt + (size_t)k * DQ_ * DQ_;
    v8f acc[4] = {{}, {}, {}, {}};
    for (int k0 = 0; k0 < DQ_; k0 += 32) {
      v16h a = load_a_frag(q_h, DQ_, row0, k0, lane);
#pragma unroll
      for (int j = 0; j < 4; j++) {
        v16h b = load_b_frag(Bt, DQ_, col0 + j * 16, k0, lane);
        acc[j] = __builtin_amdgcn_wmma_f32_16x16x32_f16(false, a, false, b,
                                                        (short)0, acc[j], false, false);
      }
    }
#pragma unroll
    for (int j = 0; j < 4; j++) {
      int col = col0 + j * 16 + l;
      float bv = dyn_b[k * DQ_ + col];
#pragma unroll
      for (int r = 0; r < 8; r++) {
        int row = row0 + r + half * 8;
        float cw = chart_w[row * KEXP + k];
        force[j][r] += cw * tanhf(acc[j][r] + bv);
      }
    }
  }
#pragma unroll
  for (int j = 0; j < 4; j++) {
    int col = col0 + j * 16 + l;
#pragma unroll
    for (int r = 0; r < 8; r++) {
      int row = row0 + r + half * 8;
      size_t idx = (size_t)row * DQ_ + col;
      float f = force[j][r] + drive[idx];
      float pv = p[idx];
      float pnew = pv + DT_F * (f - DAMP_F * pv);
      pn[idx] = pnew;
      qn[idx] = q[idx] + DT_F * pnew;
    }
  }
}

// ---------------------------------------------------------------------------
// Router: logits -> softmax -> top2 -> renormalized sparse chart_w. 1 thread/row.
// ---------------------------------------------------------------------------
__global__ void router_kernel(const float* __restrict__ q, const int* __restrict__ modes,
                              const int* __restrict__ op_ids, const int* __restrict__ out_start,
                              const float* __restrict__ op_embed,
                              const float* __restrict__ mode_embed,
                              const float* __restrict__ pos_embed,
                              const float* __restrict__ wq, const float* __restrict__ wc,
                              const float* __restrict__ rb,
                              float* __restrict__ chart_w, int t) {
  int b = blockIdx.x * blockDim.x + threadIdx.x;
  if (b >= B_) return;
  int mode = modes[b * T_ + t];
  int op = op_ids[b];
  int pos = t - (out_start[b] - 1);
  pos = pos < 0 ? 0 : (pos > NPOS - 1 ? NPOS - 1 : pos);
  float lr[KEXP];
#pragma unroll
  for (int k = 0; k < KEXP; k++) lr[k] = rb[k];
  for (int c = 0; c < CTX_; c++) {
    float rc = tanhf(tanhf(op_embed[op * CTX_ + c]) + mode_embed[mode * CTX_ + c]);
    rc = tanhf(rc + tanhf(pos_embed[pos * CTX_ + c]));
#pragma unroll
    for (int k = 0; k < KEXP; k++) lr[k] += rc * wc[c * KEXP + k];
  }
  for (int d = 0; d < DQ_; d++) {
    float qd = q[b * DQ_ + d];
#pragma unroll
    for (int k = 0; k < KEXP; k++) lr[k] += qd * wq[d * KEXP + k];
  }
  float mx = lr[0];
#pragma unroll
  for (int k = 1; k < KEXP; k++) mx = fmaxf(mx, lr[k]);
  float wv[KEXP], s = 0.f;
#pragma unroll
  for (int k = 0; k < KEXP; k++) { wv[k] = expf(lr[k] - mx); s += wv[k]; }
#pragma unroll
  for (int k = 0; k < KEXP; k++) wv[k] /= s;
  int i1 = 0;
#pragma unroll
  for (int k = 1; k < KEXP; k++) if (wv[k] > wv[i1]) i1 = k;
  int i2 = (i1 == 0) ? 1 : 0;
#pragma unroll
  for (int k = 0; k < KEXP; k++) if (k != i1 && wv[k] > wv[i2]) i2 = k;
  float denom = fmaxf(wv[i1] + wv[i2], 1e-8f);
#pragma unroll
  for (int k = 0; k < KEXP; k++) chart_w[b * KEXP + k] = 0.f;
  chart_w[b * KEXP + i1] = wv[i1] / denom;
  chart_w[b * KEXP + i2] = wv[i2] / denom;
}

// ---------------------------------------------------------------------------
// drive u = tanh((1-mf)*embed_in[tok] + mf*embed_out[tok])
// ---------------------------------------------------------------------------
__global__ void drive_kernel(const int* __restrict__ tokens, const int* __restrict__ modes,
                             const float* __restrict__ embed_in,
                             const float* __restrict__ embed_out,
                             float* __restrict__ drive, int t) {
  int i = blockIdx.x * blockDim.x + threadIdx.x;
  if (i >= B_ * DQ_) return;
  int b = i / DQ_, d = i % DQ_;
  int tok = tokens[b * T_ + t];
  float mf = (float)modes[b * T_ + t];
  drive[i] = tanhf((1.f - mf) * embed_in[(size_t)tok * DQ_ + d] +
                   mf * embed_out[(size_t)tok * DQ_ + d]);
}

__global__ void cvt16_kernel(const float* __restrict__ src, f16* __restrict__ dst, int n) {
  int i = blockIdx.x * blockDim.x + threadIdx.x;
  if (i < n) dst[i] = (f16)src[i];
}

// state = [q,p], energy = 0.5*sum(p^2); x = [state, energy, zero-pad] in f16
__global__ void state_kernel(const float* __restrict__ q, const float* __restrict__ p,
                             f16* __restrict__ state_h, f16* __restrict__ x_h) {
  int b = blockIdx.x, tid = threadIdx.x;  // 256 threads
  __shared__ float red[256];
  float qv = q[b * DQ_ + tid];
  float pv = p[b * DQ_ + tid];
  state_h[b * DS_ + tid] = (f16)qv;
  state_h[b * DS_ + DQ_ + tid] = (f16)pv;
  x_h[(size_t)b * XDP + tid] = (f16)qv;
  x_h[(size_t)b * XDP + DQ_ + tid] = (f16)pv;
  red[tid] = pv * pv;
  __syncthreads();
  for (int s = 128; s > 0; s >>= 1) {
    if (tid < s) red[tid] += red[tid + s];
    __syncthreads();
  }
  if (tid == 0) x_h[(size_t)b * XDP + XD_ - 1] = (f16)(0.5f * red[0]);
  if (tid < XDP - XD_) x_h[(size_t)b * XDP + XD_ + tid] = (f16)0.f;
}

// memrowk = memrow + mempos[t]
__global__ void addpos_kernel(const f16* __restrict__ memrow_h,
                              const float* __restrict__ mempos,
                              f16* __restrict__ memrowk_h, int t) {
  int i = blockIdx.x * blockDim.x + threadIdx.x;
  if (i >= B_ * DQM_) return;
  int d = i % DQM_;
  memrowk_h[i] = (f16)((float)memrow_h[i] + mempos[t * DQM_ + d]);
}

// ---------------------------------------------------------------------------
// attention over 32 memory slots + tanh + layernorm. one block per batch row.
// ---------------------------------------------------------------------------
__global__ void attn_kernel(const float* __restrict__ memK, const float* __restrict__ memV,
                            const float* __restrict__ qv, const int* __restrict__ out_start,
                            const float* __restrict__ ln_g, const float* __restrict__ ln_b,
                            f16* __restrict__ hn_h) {
  int b = blockIdx.x, tid = threadIdx.x;  // 256 threads
  __shared__ float red[256];
  __shared__ float qs[DQM_];
  __shared__ float attn[MAXMEM];
  __shared__ float mean_s, var_s;
  qs[tid] = qv[b * DQM_ + tid];
  __syncthreads();
  int slot = tid >> 3, part = tid & 7;
  const float* kr = memK + ((size_t)b * MAXMEM + slot) * DQM_;
  float s = 0.f;
  for (int d = part; d < DQM_; d += 8) s += kr[d] * qs[d];
  red[tid] = s;
  __syncthreads();
  if (part == 0) {
    float tot = 0.f;
    for (int j = 0; j < 8; j++) tot += red[slot * 8 + j];
    int mem_len = out_start[b] - 4;
    mem_len = mem_len < 0 ? 0 : (mem_len > MAXMEM ? MAXMEM : mem_len);
    attn[slot] = (slot < mem_len) ? tot * SCALE_F : -1e9f;
  }
  __syncthreads();
  if (tid == 0) {
    float mx = -1e30f;
    for (int m = 0; m < MAXMEM; m++) mx = fmaxf(mx, attn[m]);
    float ss = 0.f;
    for (int m = 0; m < MAXMEM; m++) { attn[m] = expf(attn[m] - mx); ss += attn[m]; }
    for (int m = 0; m < MAXMEM; m++) attn[m] /= ss;
  }
  __syncthreads();
  float ctx = 0.f;
  for (int m = 0; m < MAXMEM; m++)
    ctx += attn[m] * memV[((size_t)b * MAXMEM + m) * DQM_ + tid];
  float h = tanhf(ctx + qs[tid]);
  red[tid] = h;
  __syncthreads();
  for (int s2 = 128; s2 > 0; s2 >>= 1) {
    if (tid < s2) red[tid] += red[tid + s2];
    __syncthreads();
  }
  if (tid == 0) mean_s = red[0] / DQM_;
  __syncthreads();
  float dv = h - mean_s;
  red[tid] = dv * dv;
  __syncthreads();
  for (int s2 = 128; s2 > 0; s2 >>= 1) {
    if (tid < s2) red[tid] += red[tid + s2];
    __syncthreads();
  }
  if (tid == 0) var_s = red[0] / DQM_;
  __syncthreads();
  float hn = dv * rsqrtf(var_s + 1e-5f) * ln_g[tid] + ln_b[tid];
  hn_h[b * DQM_ + tid] = (f16)hn;
}

// ---------------------------------------------------------------------------
// One-time weight conversions (run every launch; deterministic)
// ---------------------------------------------------------------------------
__global__ void cvt_readout_kernel(const float* __restrict__ src, f16* __restrict__ dst) {
  size_t i = (size_t)blockIdx.x * blockDim.x + threadIdx.x;
  size_t total = (size_t)KEXP * V_ * XDP;
  if (i >= total) return;
  int d = (int)(i % XDP);
  size_t kv = i / XDP;
  dst[i] = (d < XD_) ? (f16)src[kv * XD_ + d] : (f16)0.f;
}

__global__ void cvt_transpose_kernel(const float* __restrict__ src, f16* __restrict__ dst,
                                     int rows, int cols) {
  int i = blockIdx.x * blockDim.x + threadIdx.x;
  if (i >= rows * cols) return;
  int r = i / cols, c = i % cols;
  dst[(size_t)c * rows + r] = (f16)src[i];
}

__global__ void cvt_dynA_kernel(const float* __restrict__ src, f16* __restrict__ dst) {
  int i = blockIdx.x * blockDim.x + threadIdx.x;
  if (i >= KEXP * DQ_ * DQ_) return;
  int k = i / (DQ_ * DQ_), rem = i % (DQ_ * DQ_);
  int d = rem / DQ_, e = rem % DQ_;
  dst[((size_t)k * DQ_ + e) * DQ_ + d] = (f16)src[i];
}

__global__ void init_qp_kernel(const float* __restrict__ init, float* __restrict__ q,
                               float* __restrict__ p) {
  int i = blockIdx.x * blockDim.x + threadIdx.x;
  if (i >= B_ * DQ_) return;
  int b = i / DQ_, d = i % DQ_;
  q[i] = init[b * DS_ + d];
  p[i] = init[b * DS_ + DQ_ + d];
}

// K/V cache init for the all-zero memory state: mem0 = tanh(mem_in_b)
__global__ void init_memkv_kernel(const float* __restrict__ mem_in_b,
                                  const float* __restrict__ mempos,
                                  const float* __restrict__ kW, const float* __restrict__ kb,
                                  const float* __restrict__ vW, const float* __restrict__ vb,
                                  float* __restrict__ memK, float* __restrict__ memV) {
  int i = blockIdx.x * blockDim.x + threadIdx.x;
  if (i >= MAXMEM * DQM_) return;
  int m = i / DQM_, d = i % DQM_;
  float k0 = kb[d], v0 = vb[d];
  for (int e = 0; e < DQM_; e++) {
    float m0 = tanhf(mem_in_b[e]);
    k0 += (m0 + mempos[m * DQM_ + e]) * kW[e * DQM_ + d];
    v0 += m0 * vW[e * DQM_ + d];
  }
  for (int b = 0; b < B_; b++) {
    memK[((size_t)b * MAXMEM + m) * DQM_ + d] = k0;
    memV[((size_t)b * MAXMEM + m) * DQM_ + d] = v0;
  }
}

// ---------------------------------------------------------------------------
extern "C" void kernel_launch(void* const* d_in, const int* in_sizes, int n_in,
                              void* d_out, int out_size, void* d_ws, size_t ws_size,
                              hipStream_t stream) {
  const int*   tokens     = (const int*)d_in[0];
  const int*   modes      = (const int*)d_in[1];
  const int*   op_ids     = (const int*)d_in[2];
  const int*   out_start  = (const int*)d_in[3];
  const float* init_state = (const float*)d_in[5];
  const float* embed_in   = (const float*)d_in[6];
  const float* embed_out  = (const float*)d_in[7];
  const float* readout_w  = (const float*)d_in[8];
  const float* readout_b  = (const float*)d_in[9];
  const float* op_embed   = (const float*)d_in[10];
  const float* mode_embed = (const float*)d_in[11];
  const float* pos_embed  = (const float*)d_in[12];
  const float* router_wq  = (const float*)d_in[13];
  const float* router_wc  = (const float*)d_in[14];
  const float* router_b   = (const float*)d_in[15];
  const float* dyn_A      = (const float*)d_in[16];
  const float* dyn_b      = (const float*)d_in[17];
  const float* mem_in_w   = (const float*)d_in[18];
  const float* mem_in_b   = (const float*)d_in[19];
  const float* query_in_w = (const float*)d_in[20];
  const float* query_in_b = (const float*)d_in[21];
  const float* q_proj_w   = (const float*)d_in[22];
  const float* q_proj_b   = (const float*)d_in[23];
  const float* k_proj_w   = (const float*)d_in[24];
  const float* k_proj_b   = (const float*)d_in[25];
  const float* v_proj_w   = (const float*)d_in[26];
  const float* v_proj_b   = (const float*)d_in[27];
  const float* mempos     = (const float*)d_in[28];
  const float* ln_g       = (const float*)d_in[29];
  const float* ln_b       = (const float*)d_in[30];
  const float* ro1_w      = (const float*)d_in[31];
  const float* ro1_b      = (const float*)d_in[32];
  const float* ro2_w      = (const float*)d_in[33];
  const float* ro2_b      = (const float*)d_in[34];
  float* out = (float*)d_out;

  char* wp = (char*)d_ws;
  auto alloc = [&](size_t bytes) -> void* {
    void* p = (void*)wp;
    wp += (bytes + 255) & ~(size_t)255;
    return p;
  };
  f16* Wro_h    = (f16*)alloc(sizeof(f16) * (size_t)KEXP * V_ * XDP);
  f16* ro2t     = (f16*)alloc(sizeof(f16) * (size_t)V_ * DQM_);
  f16* ro1t     = (f16*)alloc(sizeof(f16) * DQM_ * DQM_);
  f16* memint   = (f16*)alloc(sizeof(f16) * DQM_ * DS_);
  f16* queryint = (f16*)alloc(sizeof(f16) * DQM_ * DS_);
  f16* qprojt   = (f16*)alloc(sizeof(f16) * DQM_ * DQM_);
  f16* kprojt   = (f16*)alloc(sizeof(f16) * DQM_ * DQM_);
  f16* vprojt   = (f16*)alloc(sizeof(f16) * DQM_ * DQM_);
  f16* dynAt    = (f16*)alloc(sizeof(f16) * KEXP * DQ_ * DQ_);
  float* q0 = (float*)alloc(sizeof(float) * B_ * DQ_);
  float* p0 = (float*)alloc(sizeof(float) * B_ * DQ_);
  float* q1 = (float*)alloc(sizeof(float) * B_ * DQ_);
  float* p1 = (float*)alloc(sizeof(float) * B_ * DQ_);
  f16* q_h = (f16*)alloc(sizeof(f16) * B_ * DQ_);
  float* chart = (float*)alloc(sizeof(float) * B_ * KEXP);
  float* drive = (float*)alloc(sizeof(float) * B_ * DQ_);
  f16* state_h = (f16*)alloc(sizeof(f16) * B_ * DS_);
  f16* x_h     = (f16*)alloc(sizeof(f16) * (size_t)B_ * XDP);
  f16* memrow_h  = (f16*)alloc(sizeof(f16) * B_ * DQM_);
  f16* memrowk_h = (f16*)alloc(sizeof(f16) * B_ * DQM_);
  f16* tmp1_h    = (f16*)alloc(sizeof(f16) * B_ * DQM_);
  float* qv_f    = (float*)alloc(sizeof(float) * B_ * DQM_);
  f16* hn_h      = (f16*)alloc(sizeof(f16) * B_ * DQM_);
  f16* r1_h      = (f16*)alloc(sizeof(f16) * B_ * DQM_);
  float* memK = (float*)alloc(sizeof(float) * (size_t)B_ * MAXMEM * DQM_);
  float* memV = (float*)alloc(sizeof(float) * (size_t)B_ * MAXMEM * DQM_);

  // ---- prologue: weight conversion + state init -------------------------
  {
    size_t total = (size_t)KEXP * V_ * XDP;
    cvt_readout_kernel<<<(unsigned)((total + 255) / 256), 256, 0, stream>>>(readout_w, Wro_h);
    cvt_transpose_kernel<<<(DQM_ * V_ + 255) / 256, 256, 0, stream>>>(ro2_w, ro2t, DQM_, V_);
    cvt_transpose_kernel<<<(DQM_ * DQM_ + 255) / 256, 256, 0, stream>>>(ro1_w, ro1t, DQM_, DQM_);
    cvt_transpose_kernel<<<(DS_ * DQM_ + 255) / 256, 256, 0, stream>>>(mem_in_w, memint, DS_, DQM_);
    cvt_transpose_kernel<<<(DS_ * DQM_ + 255) / 256, 256, 0, stream>>>(query_in_w, queryint, DS_, DQM_);
    cvt_transpose_kernel<<<(DQM_ * DQM_ + 255) / 256, 256, 0, stream>>>(q_proj_w, qprojt, DQM_, DQM_);
    cvt_transpose_kernel<<<(DQM_ * DQM_ + 255) / 256, 256, 0, stream>>>(k_proj_w, kprojt, DQM_, DQM_);
    cvt_transpose_kernel<<<(DQM_ * DQM_ + 255) / 256, 256, 0, stream>>>(v_proj_w, vprojt, DQM_, DQM_);
    cvt_dynA_kernel<<<(KEXP * DQ_ * DQ_ + 255) / 256, 256, 0, stream>>>(dyn_A, dynAt);
    init_qp_kernel<<<(B_ * DQ_ + 255) / 256, 256, 0, stream>>>(init_state, q0, p0);
    init_memkv_kernel<<<(MAXMEM * DQM_ + 255) / 256, 256, 0, stream>>>(
        mem_in_b, mempos, k_proj_w, k_proj_b, v_proj_w, v_proj_b, memK, memV);
  }

  // 16x64 tiles per wave, 8 waves per block
  const int smallGemmBlocks = (B_ / 16) * (DQM_ / 64) / 8;  // 4
  const int bigGemmBlocks   = (B_ / 16) * (V_ / 64) / 8;    // 128
  const int forceBlocks     = (B_ / 16) * (DQ_ / 64) / 8;   // 4

  for (int t = 0; t < TM1; t++) {
    drive_kernel<<<(B_ * DQ_ + 255) / 256, 256, 0, stream>>>(tokens, modes, embed_in,
                                                             embed_out, drive, t);
    // EVO iteration 0: (q0,p0) -> (q1,p1)
    cvt16_kernel<<<(B_ * DQ_ + 255) / 256, 256, 0, stream>>>(q0, q_h, B_ * DQ_);
    router_kernel<<<1, 128, 0, stream>>>(q0, modes, op_ids, out_start, op_embed,
                                         mode_embed, pos_embed, router_wq, router_wc,
                                         router_b, chart, t);
    force_kernel<<<forceBlocks, 256, 0, stream>>>(q_h, q0, p0, dynAt, dyn_b,
                                                  chart, drive, q1, p1);
    // EVO iteration 1: (q1,p1) -> (q0,p0); chart_w from this iteration is used below
    cvt16_kernel<<<(B_ * DQ_ + 255) / 256, 256, 0, stream>>>(q1, q_h, B_ * DQ_);
    router_kernel<<<1, 128, 0, stream>>>(q1, modes, op_ids, out_start, op_embed,
                                         mode_embed, pos_embed, router_wq, router_wc,
                                         router_b, chart, t);
    force_kernel<<<forceBlocks, 256, 0, stream>>>(q_h, q1, p1, dynAt, dyn_b,
                                                  chart, drive, q0, p0);
    // state / x
    state_kernel<<<B_, 256, 0, stream>>>(q0, p0, state_h, x_h);
    // expert readout -> logits (write)
    readout_kernel<<<bigGemmBlocks, 256, 0, stream>>>(x_h, Wro_h, readout_b, chart,
                                                      out + (size_t)t * V_, TM1 * V_);
    // incremental memory-bank K/V cache update (only slots 0..31 ever change)
    if (t < MAXMEM) {
      gemm_kernel<<<smallGemmBlocks, 256, 0, stream>>>(state_h, memint, mem_in_b,
          nullptr, memrow_h, B_, DQM_, DS_, DS_, DS_, DQM_, 1, 0);
      addpos_kernel<<<(B_ * DQM_ + 255) / 256, 256, 0, stream>>>(memrow_h, mempos,
                                                                 memrowk_h, t);
      gemm_kernel<<<smallGemmBlocks, 256, 0, stream>>>(memrowk_h, kprojt, k_proj_b,
          memK + t * DQM_, nullptr, B_, DQM_, DQM_, DQM_, DQM_, MAXMEM * DQM_, 0, 0);
      gemm_kernel<<<smallGemmBlocks, 256, 0, stream>>>(memrow_h, vprojt, v_proj_b,
          memV + t * DQM_, nullptr, B_, DQM_, DQM_, DQM_, DQM_, MAXMEM * DQM_, 0, 0);
    }
    // query path
    gemm_kernel<<<smallGemmBlocks, 256, 0, stream>>>(state_h, queryint, query_in_b,
        nullptr, tmp1_h, B_, DQM_, DS_, DS_, DS_, DQM_, 1, 0);
    gemm_kernel<<<smallGemmBlocks, 256, 0, stream>>>(tmp1_h, qprojt, q_proj_b,
        qv_f, nullptr, B_, DQM_, DQM_, DQM_, DQM_, DQM_, 1, 0);
    // attention + LN
    attn_kernel<<<B_, 256, 0, stream>>>(memK, memV, qv_f, out_start, ln_g, ln_b, hn_h);
    // readout MLP: ro1 (tanh) then ro2 accumulated into logits
    gemm_kernel<<<smallGemmBlocks, 256, 0, stream>>>(hn_h, ro1t, ro1_b,
        nullptr, r1_h, B_, DQM_, DQM_, DQM_, DQM_, DQM_, 1, 0);
    gemm_kernel<<<bigGemmBlocks, 256, 0, stream>>>(r1_h, ro2t, ro2_b,
        out + (size_t)t * V_, nullptr, B_, V_, DQM_, DQM_, DQM_, TM1 * V_, 0, 1);
  }
}
